// Sae_84378927497417
// MI455X (gfx1250) — matrix operand order repaired and checked by reading, not compiled, not run on hardware
//
#include <hip/hip_runtime.h>

// ---------------------------------------------------------------------------
// SAE forward, fused: encode (bf16 WMMA, f32 accum) -> top-20 -> sparse decode
// gfx1250 (CDNA5), wave32. 32 rows/block: each B fragment feeds 2 M-tiles,
// halving L2 traffic for the streamed W_enc. x-tile staged into LDS with
// global_load_async_to_lds_b128 (ASYNCcnt). No software prefetch: the 8-WMMA
// loop body + compiler partial waits already cover L0 miss latency, and
// inline-asm prefetch breaks clause formation / adds hazard NOPs.
// ---------------------------------------------------------------------------

typedef __attribute__((ext_vector_type(16))) __bf16        v16bf;
typedef __attribute__((ext_vector_type(8)))  float         v8f;
typedef __attribute__((ext_vector_type(8)))  unsigned int  v8ui;

#define D_MODEL   768
#define D_LATENT  12288
#define N_ROWS    16384
#define TOPK      20
#define TILE_ROWS 32                   // 2 M-tiles per block
#define CHUNK     512
#define NCHUNK    (D_LATENT / CHUNK)   // 24
#define KSTEPS    (D_MODEL / 32)       // 24
#define AST       776                  // padded bf16 elements per abuf row (1552 B, 16B aligned)
#define CST       516                  // padded fp32 elements per cbuf row (8*516 % 64 == 32)

// dynamic LDS layout (bytes)
#define OFF_ABUF  0
#define OFF_CBUF  (TILE_ROWS * AST * 2)               // 49664
#define OFF_TOPV  (OFF_CBUF + TILE_ROWS * CST * 4)    // 115712
#define OFF_TOPI  (OFF_TOPV + 256 * TOPK * 4)         // 136192
#define OFF_MVAL  (OFF_TOPI + 256 * TOPK * 2)         // 146432
#define OFF_MIDX  (OFF_MVAL + TILE_ROWS * TOPK * 4)   // 148992
#define OFF_HD    (OFF_MIDX + TILE_ROWS * TOPK * 4)   // 151552
#define SMEM_BYTES (OFF_HD + 256)                     // 151808  (< 160KB)

__device__ __forceinline__ unsigned short f2bf(float f) {
    unsigned u = __builtin_bit_cast(unsigned, f);
    unsigned r = u + 0x7FFFu + ((u >> 16) & 1u);   // round-to-nearest-even
    return (unsigned short)(r >> 16);
}

__device__ __forceinline__ v16bf pack_frag(uint4 a, uint4 b) {
    v8ui r;
    r[0] = a.x; r[1] = a.y; r[2] = a.z; r[3] = a.w;
    r[4] = b.x; r[5] = b.y; r[6] = b.z; r[7] = b.w;
    return __builtin_bit_cast(v16bf, r);
}

// ---- pre-pass: fp32 -> bf16, vectorized 8 elements/thread ----
__global__ __launch_bounds__(256) void convert_bf16(
    const float* __restrict__ w, unsigned short* __restrict__ o, int n8) {
    int i = blockIdx.x * blockDim.x + threadIdx.x;
    if (i >= n8) return;
    const float4* p = (const float4*)w + (size_t)i * 2;
    float4 a = p[0], b = p[1];
    uint4 d;
    d.x = (unsigned)f2bf(a.x) | ((unsigned)f2bf(a.y) << 16);
    d.y = (unsigned)f2bf(a.z) | ((unsigned)f2bf(a.w) << 16);
    d.z = (unsigned)f2bf(b.x) | ((unsigned)f2bf(b.y) << 16);
    d.w = (unsigned)f2bf(b.z) | ((unsigned)f2bf(b.w) << 16);
    ((uint4*)o)[i] = d;
}

// ---- fused SAE kernel: one block = 32 rows of x ----
__global__ __launch_bounds__(256) void sae_fused(
    const unsigned short* __restrict__ xb,     // [16384, 768] bf16 (workspace)
    const unsigned short* __restrict__ wencb,  // [12288, 768] bf16 (workspace)
    const float*          __restrict__ b_enc,  // [12288]
    const float*          __restrict__ w_dec,  // [768, 12288] fp32
    const float*          __restrict__ b_dec,  // [768]
    float*                __restrict__ out)    // [16384, 768]
{
    extern __shared__ unsigned char smem[];
    unsigned short* abuf = (unsigned short*)(smem + OFF_ABUF);
    float*          cbuf = (float*)(smem + OFF_CBUF);
    float*          topv = (float*)(smem + OFF_TOPV);
    unsigned short* topi = (unsigned short*)(smem + OFF_TOPI);
    float*          mval = (float*)(smem + OFF_MVAL);
    int*            midx = (int*)(smem + OFF_MIDX);
    unsigned char*  hd   = (unsigned char*)(smem + OFF_HD);

    const int tid      = threadIdx.x;
    const int row_base = blockIdx.x * TILE_ROWS;
    const int srow     = tid >> 3;   // 0..31 : row handled in scan/decode phases
    const int s8       = tid & 7;    // 0..7  : slot within the row

    // per-thread top-k sublist state (persists across chunks)
    float*          lv = topv + tid * TOPK;
    unsigned short* li = topi + tid * TOPK;
    float           minv = -3.0e38f;           // register copy of lv[TOPK-1]

    // ---- stage x tile (32 x 768 bf16) into LDS via async copy; init lists ----
    {
        const unsigned abase = (unsigned)(size_t)(void*)abuf;  // LDS byte offset
        const unsigned lbase = abase + (unsigned)(srow * (AST * 2) + s8 * 16);
        const unsigned short* gx =
            xb + (size_t)(row_base + srow) * D_MODEL + s8 * 8;
        #pragma unroll
        for (int p = 0; p < 12; ++p) {                 // 12 x 16B per thread
            unsigned lds_off = lbase + (unsigned)(p * 128);
            unsigned long long ga = (unsigned long long)(size_t)(gx + p * 64);
            asm volatile("global_load_async_to_lds_b128 %0, %1, off"
                         :: "v"(lds_off), "v"(ga));
        }
        #pragma unroll
        for (int k = 0; k < TOPK; ++k) { lv[k] = -3.0e38f; li[k] = 0; }
        asm volatile("s_wait_asynccnt 0x0" ::: "memory");
    }
    __syncthreads();

    const int wave = tid >> 5;        // 0..7
    const int lane = tid & 31;
    const int half = lane >> 4;       // lane-half select per ISA fragment layout
    const int hm   = lane & 15;
    const float* crow = cbuf + srow * CST;

    for (int cc = 0; cc < NCHUNK; ++cc) {
        const int cbase = cc * CHUNK;

        // ------- encode: 2 M-tiles x 4 N-tiles per wave (8 WMMA / k-step) -------
        v8f acc[2][4] = {};
        const int l0 = cbase + wave * 64 + hm;     // W_enc row for N-tile t: l0 + 16t

        for (int ks = 0; ks < KSTEPS; ++ks) {
            const int kb = ks * 32;
            // A fragments (16x32 bf16) from LDS, one per M-tile:
            //   lanes 0-15: K = kb+{0..15}; lanes 16-31: K = kb+{16..31}
            v16bf af[2];
            #pragma unroll
            for (int mt = 0; mt < 2; ++mt) {
                const uint4* ap0 =
                    (const uint4*)(abuf + (mt * 16 + hm) * AST + kb + half * 16);
                const uint4* ap1 =
                    (const uint4*)(abuf + (mt * 16 + hm) * AST + kb + 8 + half * 16);
                af[mt] = pack_frag(ap0[0], ap1[0]);
            }
            #pragma unroll
            for (int t = 0; t < 4; ++t) {
                // B fragment (32x16 bf16): lane n = column n = W_enc row l0+16t,
                // 16 contiguous K at kb + half*16. One B feeds BOTH M-tiles.
                const unsigned short* wp =
                    wencb + (size_t)(l0 + t * 16) * D_MODEL + kb + half * 16;
                uint4 b0 = ((const uint4*)wp)[0];
                uint4 b1 = ((const uint4*)wp)[1];
                v16bf bfrag = pack_frag(b0, b1);
                acc[0][t] = __builtin_amdgcn_wmma_f32_16x16x32_bf16(
                    false, af[0], false, bfrag, (short)0, acc[0][t], false, false);
                acc[1][t] = __builtin_amdgcn_wmma_f32_16x16x32_bf16(
                    false, af[1], false, bfrag, (short)0, acc[1][t], false, false);
            }
        }

        // C layout: VGPR r -> (M = half*8 + r, N = hm); add b_enc, spill to LDS
        #pragma unroll
        for (int t = 0; t < 4; ++t) {
            const int col = wave * 64 + t * 16 + hm;
            const float be = b_enc[cbase + col];
            #pragma unroll
            for (int mt = 0; mt < 2; ++mt) {
                #pragma unroll
                for (int r = 0; r < 8; ++r) {
                    cbuf[(mt * 16 + half * 8 + r) * CST + col] = acc[mt][t][r] + be;
                }
            }
        }
        __syncthreads();

        // ------- top-k scan: 8 threads per row; hot path = ds_load + cmp -------
        for (int j = 0; j < CHUNK / 8; ++j) {
            const int   c = s8 + (j << 3);
            const float v = crow[c];
            if (v > minv) {                            // rare: insertion
                const int gidx = cbase + c;
                int p = TOPK - 1;
                while (p > 0 && lv[p - 1] < v) {
                    lv[p] = lv[p - 1]; li[p] = li[p - 1]; --p;
                }
                lv[p] = v; li[p] = (unsigned short)gidx;
                minv = lv[TOPK - 1];
            }
        }
        __syncthreads();   // protect cbuf before next chunk overwrites it
    }

    // ------- merge 8 sorted sublists per row (threads 0..31, one row each) -------
    if (tid < TILE_ROWS) {
        unsigned char* h = hd + tid * 8;
        #pragma unroll
        for (int j = 0; j < 8; ++j) h[j] = 0;
        for (int k = 0; k < TOPK; ++k) {
            float best = -3.4e38f; int bj = 0;
            #pragma unroll
            for (int j = 0; j < 8; ++j) {
                const int hj = h[j];
                const float v = topv[(tid * 8 + j) * TOPK + hj];
                if (v > best) { best = v; bj = j; }
            }
            const int hb = h[bj];
            mval[tid * TOPK + k] = best;
            midx[tid * TOPK + k] = (int)topi[(tid * 8 + bj) * TOPK + hb];
            h[bj] = (unsigned char)(hb + 1);
        }
    }
    __syncthreads();

    // ------- sparse decode: out = sum_k val_k * W_dec[:, idx_k] + b_dec -------
    {
        const size_t orow = (size_t)(row_base + srow) * D_MODEL;
        // preload this row's top-k (val, idx) into registers once
        float rv[TOPK]; int ri[TOPK];
        #pragma unroll
        for (int k = 0; k < TOPK; ++k) {
            rv[k] = mval[srow * TOPK + k];
            ri[k] = midx[srow * TOPK + k];
        }
        for (int j = 0; j < D_MODEL / 8; ++j) {    // 96 output elements per thread
            const int c = s8 + (j << 3);
            float accd = b_dec[c];
            const float* wrow = w_dec + (size_t)c * D_LATENT;
            #pragma unroll
            for (int k = 0; k < TOPK; ++k) {
                accd += rv[k] * wrow[ri[k]];       // gathered from L2-resident W_dec
            }
            out[orow + c] = accd;
        }
    }
}

// ---------------------------------------------------------------------------
extern "C" void kernel_launch(void* const* d_in, const int* in_sizes, int n_in,
                              void* d_out, int out_size, void* d_ws, size_t ws_size,
                              hipStream_t stream) {
    const float* x    = (const float*)d_in[0];   // [16384, 768]
    const float* Wenc = (const float*)d_in[1];   // [12288, 768]
    const float* benc = (const float*)d_in[2];   // [12288]
    const float* Wdec = (const float*)d_in[3];   // [768, 12288]
    const float* bdec = (const float*)d_in[4];   // [768]
    float* out = (float*)d_out;

    unsigned short* wencb = (unsigned short*)d_ws;               // 18.9 MB
    unsigned short* xb    = (unsigned short*)((char*)d_ws + (size_t)D_LATENT * D_MODEL * 2);

    const int n8w = (D_LATENT * D_MODEL) / 8;    // 1,179,648 -> 4608 blocks
    const int n8x = (N_ROWS   * D_MODEL) / 8;    // 1,572,864 -> 6144 blocks
    convert_bf16<<<n8w / 256, 256, 0, stream>>>(Wenc, wencb, n8w);
    convert_bf16<<<n8x / 256, 256, 0, stream>>>(x,    xb,    n8x);

    sae_fused<<<N_ROWS / TILE_ROWS, 256, SMEM_BYTES, stream>>>(
        xb, wencb, benc, Wdec, bdec, out);
}
